// VectorQuantizer_82463372083584
// MI455X (gfx1250) — matrix operand order repaired and checked by reading, not compiled, not run on hardware
//
#include <hip/hip_runtime.h>
#include <hip/hip_bf16.h>

// ---------------------------------------------------------------------------
// VQ-VAE codebook lookup for MI455X (gfx1250, wave32, WMMA).
//   inputs:      [32, 512, 32, 32] f32 (NCHW)   -> N = 32768 rows, D = 512
//   embeddings:  [4096, 512] f32                -> K = 4096 codes
//   out: quantized NCHW f32 (16777216 floats) ++ loss (1 float)
//
// Score GEMM (137 GFLOP) on v_wmma_f32_16x16x32_bf16, f32 accumulate.
// argmin(||z-e||^2) == argmin(||e||^2 - 2 z.e); -2 folded into bf16(z),
// ||e||^2 pre-seeded into the WMMA C operand. Tile staging uses the CDNA5
// async global->LDS path (ASYNCcnt-tracked) confirmed present on this
// toolchain.
// ---------------------------------------------------------------------------

typedef __bf16 v16bf __attribute__((ext_vector_type(16)));
typedef float  v8f   __attribute__((ext_vector_type(8)));

#define DDIM   512
#define KCB    4096
#define NROWS  32768
#define MT     64          // rows per block in argmin kernel
#define NCH    64          // score-columns per chunk
#define LDA    520         // padded LDS stride (bf16 elems): breaks bank conflicts
#define LDB    520

union Frag16 { uint4 q[2]; v16bf v; };

// ---- CDNA5 async global->LDS (direct, ASYNCcnt-tracked) -------------------
// Probe round 2: builtin exists; param 0 is AS1 pointer to
// 'int __attribute__((vector_size(16)))' (non-const).
typedef int v4i __attribute__((vector_size(16)));
typedef __attribute__((address_space(1))) v4i* gv4p;   // global
typedef __attribute__((address_space(3))) v4i* lv4p;   // LDS

#if __has_builtin(__builtin_amdgcn_global_load_async_to_lds_b128)
#define HAVE_ASYNC_LDS 1
__device__ __forceinline__ void async_ld16(void* lds, const void* g) {
    __builtin_amdgcn_global_load_async_to_lds_b128(
        (gv4p)(void*)g, (lv4p)lds, 0, 0);
}
#else
__device__ __forceinline__ void async_ld16(void* lds, const void* g) {
    *(uint4*)lds = *(const uint4*)g;   // fallback: VGPR round trip
}
#endif

__device__ __forceinline__ void wait_async() {
#if __has_builtin(__builtin_amdgcn_s_wait_asynccnt)
    __builtin_amdgcn_s_wait_asynccnt(0);
#elif defined(HAVE_ASYNC_LDS)
    asm volatile("s_wait_asynccnt 0x0" ::: "memory");
#endif
}

// ---------------------------------------------------------------------------
// Stage 1a: NCHW f32 -> (N, D) bf16, scaled by -2.  LDS-tiled transpose so
// global reads (along HW) and writes (along D) both coalesce.
// grid = 32 (B) * 16 (D/32) * 32 (HW/32) = 16384 blocks of 256.
// ---------------------------------------------------------------------------
__global__ __launch_bounds__(256) void k_zt(const float* __restrict__ in,
                                            __bf16* __restrict__ zbf) {
    __shared__ float tile[32][33];
    int blk = blockIdx.x;
    int b   = blk >> 9;
    int rem = blk & 511;
    int d0  = (rem >> 5) << 5;
    int hw0 = (rem & 31) << 5;
    int t = threadIdx.x;
    int c = t & 31, rbase = t >> 5;
    const size_t inb = (size_t)b * DDIM * 1024;
#pragma unroll
    for (int i = 0; i < 4; ++i) {
        int dd = rbase + i * 8;
        tile[dd][c] = in[inb + (size_t)(d0 + dd) * 1024 + hw0 + c];
    }
    __syncthreads();
#pragma unroll
    for (int i = 0; i < 4; ++i) {
        int ww = rbase + i * 8;
        zbf[((size_t)b * 1024 + hw0 + ww) * DDIM + d0 + c] =
            (__bf16)(-2.0f * tile[c][ww]);
    }
}

// ---------------------------------------------------------------------------
// Stage 1b: embeddings f32 -> bf16 + per-row squared norm. One wave per row.
// ---------------------------------------------------------------------------
__global__ __launch_bounds__(256) void k_eprep(const float* __restrict__ emb,
                                               __bf16* __restrict__ ebf,
                                               float* __restrict__ enorm) {
    int row  = (blockIdx.x * 256 + threadIdx.x) >> 5;
    int lane = threadIdx.x & 31;
    const float* src = emb + (size_t)row * DDIM;
    __bf16*      dst = ebf + (size_t)row * DDIM;
    float acc = 0.0f;
#pragma unroll
    for (int j = 0; j < 16; ++j) {
        float e = src[lane + 32 * j];
        acc += e * e;
        dst[lane + 32 * j] = (__bf16)e;
    }
#pragma unroll
    for (int m = 16; m >= 1; m >>= 1) acc += __shfl_xor(acc, m, 32);
    if (lane == 0) enorm[row] = acc;
}

// ---------------------------------------------------------------------------
// Stage 2: WMMA argmin.  Block = 128 threads = 4 waves arranged 2x2:
//   wave (mw, nw) computes output patch rows [mw*32, +32) x cols [nw*32, +32)
//   => 2 A-frags x 2 B-frags -> 4 WMMAs per K-step (2.0 ds_load per wmma).
// A tile 64x512 bf16 + B chunk 64x512 bf16 staged via async global->LDS.
// ---------------------------------------------------------------------------
__global__ __launch_bounds__(128) void k_argmin(const __bf16* __restrict__ zbf,
                                                const __bf16* __restrict__ ebf,
                                                const float* __restrict__ enorm,
                                                int* __restrict__ outIdx) {
    __shared__ __bf16 sa[MT * LDA];
    __shared__ __bf16 sb[NCH * LDB];
    __shared__ float  redV[2][MT];
    __shared__ int    redI[2][MT];

    const int t     = threadIdx.x;
    const int n0blk = blockIdx.x * MT;

    // --- stage A tile: 64 rows x 512 bf16, each thread 32 x b128 ----------
    {
        int row = t >> 1, off = (t & 1) * 256;
        const __bf16* g = zbf + (size_t)(n0blk + row) * DDIM + off;
        __bf16*       l = sa + row * LDA + off;
#pragma unroll
        for (int i = 0; i < 32; ++i) async_ld16(l + i * 8, g + i * 8);
    }

    const int lane = t & 31;
    const int lo   = lane & 15, hi = lane >> 4;
    const int wv   = t >> 5;
    const int mw   = wv >> 1, nw = wv & 1;
    const int m0   = mw * 32;           // row base of this wave's patch
    const int nb   = nw * 32;           // col base of this wave's patch
    const int ab0  = (m0 + lo) * LDA + 8 * hi;
    const int ab1  = (m0 + 16 + lo) * LDA + 8 * hi;
    const int bb0  = (nb + lo) * LDB + 8 * hi;
    const int bb1  = (nb + 16 + lo) * LDB + 8 * hi;

    float bestV[2][8];
    int   bestI[2][8];
#pragma unroll
    for (int i = 0; i < 2; ++i)
#pragma unroll
        for (int r = 0; r < 8; ++r) { bestV[i][r] = 3.4e38f; bestI[i][r] = 0; }

    for (int kk = 0; kk < KCB; kk += NCH) {
        __syncthreads();                // all reads of sb from prev chunk done
        // --- stage B chunk: 64 code rows x 512 bf16, async to LDS ---------
        {
            int row = t >> 1, off = (t & 1) * 256;
            const __bf16* g = ebf + (size_t)(kk + row) * DDIM + off;
            __bf16*       l = sb + row * LDB + off;
#pragma unroll
            for (int i = 0; i < 32; ++i) async_ld16(l + i * 8, g + i * 8);
            if (kk + NCH < KCB)         // pull next chunk toward L2
                __builtin_prefetch(ebf + (size_t)(kk + NCH + row) * DDIM + off, 0, 1);
        }
        wait_async();                   // my async writes to LDS landed
        __syncthreads();                // everyone's landed

        // seed accumulators with ||e||^2 (column-only -> broadcast over rows)
        v8f a00, a01, a10, a11;
        const float en0 = enorm[kk + nb + lo];
        const float en1 = enorm[kk + nb + 16 + lo];
#pragma unroll
        for (int r = 0; r < 8; ++r) { a00[r] = en0; a01[r] = en1; a10[r] = en0; a11[r] = en1; }

        // D-loop: 16 K-steps of 32; 4 WMMAs per step on a 32x32 patch
        for (int d = 0; d < DDIM; d += 32) {
            Frag16 fa0, fa1, fb0, fb1;
            fa0.q[0] = *(const uint4*)(sa + ab0 + d);
            fa0.q[1] = *(const uint4*)(sa + ab0 + d + 16);
            fa1.q[0] = *(const uint4*)(sa + ab1 + d);
            fa1.q[1] = *(const uint4*)(sa + ab1 + d + 16);
            fb0.q[0] = *(const uint4*)(sb + bb0 + d);
            fb0.q[1] = *(const uint4*)(sb + bb0 + d + 16);
            fb1.q[0] = *(const uint4*)(sb + bb1 + d);
            fb1.q[1] = *(const uint4*)(sb + bb1 + d + 16);
            a00 = __builtin_amdgcn_wmma_f32_16x16x32_bf16(
                false, fa0.v, false, fb0.v, (short)0, a00, false, false);
            a01 = __builtin_amdgcn_wmma_f32_16x16x32_bf16(
                false, fa0.v, false, fb1.v, (short)0, a01, false, false);
            a10 = __builtin_amdgcn_wmma_f32_16x16x32_bf16(
                false, fa1.v, false, fb0.v, (short)0, a10, false, false);
            a11 = __builtin_amdgcn_wmma_f32_16x16x32_bf16(
                false, fa1.v, false, fb1.v, (short)0, a11, false, false);
        }

        // running argmin over this chunk's 32 columns of the wave's patch
        const int c0 = kk + nb + lo;
        const int c1 = kk + nb + 16 + lo;
#pragma unroll
        for (int r = 0; r < 8; ++r) {
            if (a00[r] < bestV[0][r]) { bestV[0][r] = a00[r]; bestI[0][r] = c0; }
            if (a01[r] < bestV[0][r]) { bestV[0][r] = a01[r]; bestI[0][r] = c1; }
            if (a10[r] < bestV[1][r]) { bestV[1][r] = a10[r]; bestI[1][r] = c0; }
            if (a11[r] < bestV[1][r]) { bestV[1][r] = a11[r]; bestI[1][r] = c1; }
        }
    }

    // 16-lane min-reduce (C layout: lane = N mod 16, VGPR r <-> row r + 8*hi),
    // tie-break on lowest index, then stash per-(nw) results for the cross-
    // N-wave combine in LDS.
#pragma unroll
    for (int i = 0; i < 2; ++i) {
#pragma unroll
        for (int r = 0; r < 8; ++r) {
            float v = bestV[i][r];
            int   c = bestI[i][r];
#pragma unroll
            for (int m = 1; m < 16; m <<= 1) {
                float ov = __shfl_xor(v, m, 16);
                int   oc = __shfl_xor(c, m, 16);
                if (ov < v || (ov == v && oc < c)) { v = ov; c = oc; }
            }
            if (lo == 0) {
                int lrow = m0 + 16 * i + r + 8 * hi;   // local row 0..63
                redV[nw][lrow] = v;
                redI[nw][lrow] = c;
            }
        }
    }
    __syncthreads();
    if (t < MT) {
        float v0 = redV[0][t], v1 = redV[1][t];
        int   i0 = redI[0][t], i1 = redI[1][t];
        bool take1 = (v1 < v0) || (v1 == v0 && i1 < i0);
        outIdx[n0blk + t] = take1 ? i1 : i0;
    }
}

// ---------------------------------------------------------------------------
// Stage 3: gather q = E[k] (f32), write NCHW output, per-block loss partial.
// ---------------------------------------------------------------------------
__global__ __launch_bounds__(256) void k_gather(const float* __restrict__ in,
                                                const float* __restrict__ emb,
                                                const int* __restrict__ idx,
                                                float* __restrict__ out,
                                                float* __restrict__ partial) {
    int blk = blockIdx.x;
    int b = blk >> 5, h = blk & 31;
    int t = threadIdx.x;
    int w = t & 31, d0 = t >> 5;
    int n = b * 1024 + h * 32 + w;
    int k = idx[n];
    const float* erow = emb + (size_t)k * DDIM;
    size_t base = (size_t)b * 524288 + h * 32 + w;
    float acc = 0.0f;
    for (int d = d0; d < DDIM; d += 8) {
        float q = erow[d];
        size_t o = base + (size_t)d * 1024;
        float z = in[o];
        out[o] = q;
        float df = z - q;
        acc += df * df;
    }
    __shared__ float red[256];
    red[t] = acc;
    __syncthreads();
    for (int s = 128; s > 0; s >>= 1) {
        if (t < s) red[t] += red[t + s];
        __syncthreads();
    }
    if (t == 0) partial[blk] = red[0];
}

// ---------------------------------------------------------------------------
// Stage 4: deterministic reduce of 1024 partials -> loss scalar.
// loss = e_loss + 0.25*commitment = 1.25 * mean((z - q)^2)
// ---------------------------------------------------------------------------
__global__ __launch_bounds__(256) void k_loss(const float* __restrict__ partial,
                                              float* __restrict__ loss) {
    int t = threadIdx.x;
    float a = partial[t] + partial[t + 256] + partial[t + 512] + partial[t + 768];
    __shared__ float red[256];
    red[t] = a;
    __syncthreads();
    for (int s = 128; s > 0; s >>= 1) {
        if (t < s) red[t] += red[t + s];
        __syncthreads();
    }
    if (t == 0) loss[0] = 1.25f * red[0] / (float)(NROWS * (double)DDIM);
}

// ---------------------------------------------------------------------------
extern "C" void kernel_launch(void* const* d_in, const int* in_sizes, int n_in,
                              void* d_out, int out_size, void* d_ws, size_t ws_size,
                              hipStream_t stream) {
    const float* inputs = (const float*)d_in[0];   // [32,512,32,32]
    const float* emb    = (const float*)d_in[1];   // [4096,512]
    float* out = (float*)d_out;                    // 16777216 floats + 1 loss

    // workspace layout (~36.2 MB)
    char* ws = (char*)d_ws;
    __bf16* zbf    = (__bf16*)(ws);                             // 33,554,432 B
    __bf16* ebf    = (__bf16*)(ws + 33554432);                  //  4,194,304 B
    float*  enorm  = (float*)(ws + 33554432 + 4194304);         //     16,384 B
    int*    idxbuf = (int*)  (ws + 33554432 + 4194304 + 16384); //    131,072 B
    float*  part   = (float*)(ws + 33554432 + 4194304 + 16384 + 131072);

    (void)in_sizes; (void)n_in; (void)out_size; (void)ws_size;

    k_zt     <<<16384, 256, 0, stream>>>(inputs, zbf);
    k_eprep  <<<512,   256, 0, stream>>>(emb, ebf, enorm);
    k_argmin <<<NROWS / MT, 128, 0, stream>>>(zbf, ebf, enorm, idxbuf);
    k_gather <<<1024,  256, 0, stream>>>(inputs, emb, idxbuf, out, part);
    k_loss   <<<1,     256, 0, stream>>>(part, out + 16777216);
}
